// SineGRU_55611236549334
// MI455X (gfx1250) — matrix operand order repaired and checked by reading, not compiled, not run on hardware
//
#include <hip/hip_runtime.h>

// ---------------------------------------------------------------------------
// Persistent GRU for MI455X (gfx1250, wave32, WMMA).
//
// 16 workgroups x 512 threads (16 waves). Each WG owns 16 batch rows and runs
// all 2048 timesteps privately (recurrence independent across batch -> no
// global sync; one workgroup barrier per step).
//
// Wave w owns hidden columns [16w, 16w+16). W_hh B-operand tiles for its
// columns stay register-resident (23 of 24 tiles = 184 VGPRs); the 24th tile
// (g=2, kk=7 -- the last one used each step) lives in LDS and is re-read per
// step with ds_load_b128 overlapped under the kk=5/6 WMMAs. This keeps peak
// live VGPRs under the gfx1250 backend's 256 cap (beyond which it spills to
// scratch -- observed in earlier rounds).
//
// Step loop: depth-2 A-tile software pipeline (ds_load_b128 pairs prefetched
// one tile ahead), 24 v_wmma_f32_16x16x32_bf16 over 3 interleaved r/z/n
// accumulator chains, LDS-table gate constants, register gate math
// (v_exp-based sigmoid/tanh), 8 ds_store_b16 of new h, 1 barrier.
// ---------------------------------------------------------------------------

typedef __attribute__((ext_vector_type(16))) __bf16 v16bf;
typedef __attribute__((ext_vector_type(8)))  __bf16 v8bf;
typedef __attribute__((ext_vector_type(8)))  float  v8f;
typedef __attribute__((ext_vector_type(4)))  float  v4f;

#define HID      256
#define TSTEPS   2048
#define NBATCH   256
#define MROWS    16            // batch rows per workgroup
#define NTHREADS 512           // 16 waves (wave32)
#define XCH      128           // x staging chunk (timesteps)
#define LROW     (HID + 8)     // bf16 row pitch: 264*2B = 528B = 132 dwords
                               // -> consecutive rows shift 4 banks (no conflicts)
#define GCPAD    12            // gconst row pitch: 48B, 16B-aligned
#define WSPAD    24            // wsp row pitch: 48B, 16B-aligned, 12 dwords

__device__ __forceinline__ float sigmoid_f(float a) {
    return 1.0f / (1.0f + __expf(-a));
}
__device__ __forceinline__ float tanh_f(float a) {
    float e2 = __expf(-2.0f * a);
    return (1.0f - e2) / (1.0f + e2);
}
__device__ __forceinline__ v16bf cat16(v8bf lo, v8bf hi) {
    return __builtin_shufflevector(lo, hi,
        0, 1, 2, 3, 4, 5, 6, 7, 8, 9, 10, 11, 12, 13, 14, 15);
}

__global__ __launch_bounds__(NTHREADS, 1)
void gru_persistent(const float* __restrict__ x,     // [B, T, 1]
                    const float* __restrict__ h0,    // [B, H]
                    const float* __restrict__ W_ih,  // [3H, 1]
                    const float* __restrict__ W_hh,  // [3H, H]
                    const float* __restrict__ b_ih,  // [3H]
                    const float* __restrict__ b_hh,  // [3H]
                    const float* __restrict__ W_ro,  // [1, H]
                    const float* __restrict__ b_ro,  // [1]
                    float* __restrict__ out)         // [B] pred ++ [B,H] h_final
{
    __shared__ __align__(16) __bf16 hbuf[2][MROWS][LROW]; // ping-pong h (bf16)
    __shared__ float xbuf[MROWS][XCH + 4];                // staged x chunk
    __shared__ __align__(16) float  gconst[HID][GCPAD];   // per-column constants
    __shared__ __align__(16) __bf16 wsp[NTHREADS][WSPAD]; // evicted tile g2,kk7

    const int tid  = threadIdx.x;
    const int wave = tid >> 5;
    const int lane = tid & 31;
    const int hl   = lane >> 4;        // half-wave (C/D M-offset 0 or 8)
    const int ln   = lane & 15;
    const int col  = wave * 16 + ln;   // this lane's hidden column (N index)
    const int b0   = blockIdx.x * MROWS;

    // ---- per-column gate constants -> LDS table ----
    // layout: {wih_r, wih_z, wih_n, brz_r, brz_z, bih_n, bhh_n, 0}
    if (tid < HID) {
        const int c = tid;
        gconst[c][0] = W_ih[0 * HID + c];
        gconst[c][1] = W_ih[1 * HID + c];
        gconst[c][2] = W_ih[2 * HID + c];
        gconst[c][3] = b_ih[0 * HID + c] + b_hh[0 * HID + c];
        gconst[c][4] = b_ih[1 * HID + c] + b_hh[1 * HID + c];
        gconst[c][5] = b_ih[2 * HID + c];
        gconst[c][6] = b_hh[2 * HID + c];
        gconst[c][7] = 0.0f;
    }

    // ---- W_hh -> WMMA B fragments ----
    // B(k,n) for n = col; lanes 0-15 hold K kb..kb+15, lanes 16-31 hold
    // K kb+16..kb+31, packed ascending. W_hh rows contiguous in K.
    // 23 tiles in registers; tile (g=2, kk=7) goes to LDS (wsp).
    v16bf wgtA[3][7];          // kk = 0..6, all gates
    v16bf wlast_r, wlast_z;    // kk = 7, gates r and z
    #pragma unroll
    for (int g = 0; g < 3; ++g) {
        const float* wrow = W_hh + (size_t)(g * HID + col) * HID;
        #pragma unroll
        for (int kk = 0; kk < 7; ++kk) {
            const float* src = wrow + kk * 32 + hl * 16;
            v16bf wv;
            #pragma unroll
            for (int i = 0; i < 16; ++i) wv[i] = (__bf16)src[i];
            wgtA[g][kk] = wv;
        }
    }
    {
        const float* r7 = W_hh + (size_t)(0 * HID + col) * HID + 7 * 32 + hl * 16;
        const float* z7 = W_hh + (size_t)(1 * HID + col) * HID + 7 * 32 + hl * 16;
        const float* n7 = W_hh + (size_t)(2 * HID + col) * HID + 7 * 32 + hl * 16;
        v16bf wr, wz;
        #pragma unroll
        for (int i = 0; i < 16; ++i) { wr[i] = (__bf16)r7[i]; wz[i] = (__bf16)z7[i]; }
        wlast_r = wr;
        wlast_z = wz;
        #pragma unroll
        for (int i = 0; i < 16; ++i) wsp[tid][i] = (__bf16)n7[i];
    }

    // ---- init hidden state in LDS ----
    for (int e = tid; e < MROWS * HID; e += NTHREADS) {
        int m = e >> 8, c = e & 255;
        hbuf[0][m][c] = (__bf16)h0[(size_t)(b0 + m) * HID + c];
    }
    __syncthreads();

    int p = 0;
    for (int t = 0; t < TSTEPS; ++t) {
        // refresh staged x every XCH steps (coalesced, amortized)
        if ((t & (XCH - 1)) == 0) {
            for (int e = tid; e < MROWS * XCH; e += NTHREADS) {
                int m = e >> 7, tt = e & (XCH - 1);
                xbuf[m][tt] = x[(size_t)(b0 + m) * TSTEPS + t + tt];
            }
            __syncthreads();
        }

        // ---- gh = h @ W_hh^T with a depth-2 A-tile pipeline ----
        // A fragment: lane -> M row = ln; two contiguous 16B K chunks.
        const __bf16* arow = &hbuf[p][ln][hl * 16];
        v8bf clo = *reinterpret_cast<const v8bf*>(arow);
        v8bf chi = *reinterpret_cast<const v8bf*>(arow + 8);
        v8f ar = {}, az = {}, an = {};
        v8bf slo = {}, shi = {};           // LDS-resident tile (g=2, kk=7)
        #pragma unroll
        for (int kk = 0; kk < 7; ++kk) {
            v8bf nlo = *reinterpret_cast<const v8bf*>(arow + (kk + 1) * 32);
            v8bf nhi = *reinterpret_cast<const v8bf*>(arow + (kk + 1) * 32 + 8);
            if (kk == 5) {   // fetch evicted weight tile; hides under kk=5/6 WMMAs
                slo = *reinterpret_cast<const v8bf*>(&wsp[tid][0]);
                shi = *reinterpret_cast<const v8bf*>(&wsp[tid][8]);
            }
            const v16bf A = cat16(clo, chi);
            ar = __builtin_amdgcn_wmma_f32_16x16x32_bf16(
                     false, A, false, wgtA[0][kk], (short)0, ar, false, false);
            az = __builtin_amdgcn_wmma_f32_16x16x32_bf16(
                     false, A, false, wgtA[1][kk], (short)0, az, false, false);
            an = __builtin_amdgcn_wmma_f32_16x16x32_bf16(
                     false, A, false, wgtA[2][kk], (short)0, an, false, false);
            clo = nlo;
            chi = nhi;
        }
        {
            const v16bf A7 = cat16(clo, chi);
            ar = __builtin_amdgcn_wmma_f32_16x16x32_bf16(
                     false, A7, false, wlast_r, (short)0, ar, false, false);
            az = __builtin_amdgcn_wmma_f32_16x16x32_bf16(
                     false, A7, false, wlast_z, (short)0, az, false, false);
            an = __builtin_amdgcn_wmma_f32_16x16x32_bf16(
                     false, A7, false, cat16(slo, shi), (short)0, an, false, false);
        }

        // ---- gate math; constants + h_old from LDS; write new h to LDS ----
        // (plain LDS loads: __syncthreads() fences keep them inside the loop)
        const v4f c0 = *reinterpret_cast<const v4f*>(&gconst[col][0]);
        const v4f c1 = *reinterpret_cast<const v4f*>(&gconst[col][4]);
        // c0 = {wih_r, wih_z, wih_n, brz_r}, c1 = {brz_z, bih_n, bhh_n, 0}

        const int q = p ^ 1;
        #pragma unroll
        for (int v = 0; v < 8; ++v) {
            const int   m  = v + 8 * hl;               // C/D fragment M row
            const float ho = (float)hbuf[p][m][col];   // previous h (bf16 state)
            const float xv = xbuf[m][t & (XCH - 1)];
            const float r  = sigmoid_f(xv * c0[0] + c0[3] + ar[v]);
            const float z  = sigmoid_f(xv * c0[1] + c1[0] + az[v]);
            const float n  = tanh_f(xv * c0[2] + c1[1] + r * (an[v] + c1[2]));
            const float hn = (1.0f - z) * n + z * ho;
            hbuf[q][m][col] = (__bf16)hn;
        }
        __syncthreads();
        p = q;

        // ---- pred = h1 @ W_ro^T + b_ro (after first step only) ----
        if (t == 0 && wave == 0 && lane < 16) {
            float s = b_ro[0];
            for (int c = 0; c < HID; ++c)
                s += (float)hbuf[p][lane][c] * W_ro[c];
            out[b0 + lane] = s;   // one-time, divergent, no WMMA inside
        }
    }

    // ---- h_final from LDS, cooperative coalesced stores ----
    for (int e = tid; e < MROWS * HID; e += NTHREADS) {
        int m = e >> 8, c = e & 255;
        out[NBATCH + (size_t)(b0 + m) * HID + c] = (float)hbuf[p][m][c];
    }
}

extern "C" void kernel_launch(void* const* d_in, const int* in_sizes, int n_in,
                              void* d_out, int out_size, void* d_ws, size_t ws_size,
                              hipStream_t stream) {
    const float* x    = (const float*)d_in[0];
    const float* h0   = (const float*)d_in[1];
    const float* W_ih = (const float*)d_in[2];
    const float* W_hh = (const float*)d_in[3];
    const float* b_ih = (const float*)d_in[4];
    const float* b_hh = (const float*)d_in[5];
    const float* W_ro = (const float*)d_in[6];
    const float* b_ro = (const float*)d_in[7];
    float* out = (float*)d_out;

    gru_persistent<<<NBATCH / MROWS, NTHREADS, 0, stream>>>(
        x, h0, W_ih, W_hh, b_ih, b_hh, W_ro, b_ro, out);
}
